// StandardSelfAttention_73495480369478
// MI455X (gfx1250) — matrix operand order repaired and checked by reading, not compiled, not run on hardware
//
#include <hip/hip_runtime.h>
#include <hip/hip_bf16.h>

// ---------------------------------------------------------------------------
// StandardSelfAttention on MI455X (gfx1250, wave32, WMMA f16 -> f32 accum)
// ---------------------------------------------------------------------------
typedef __attribute__((ext_vector_type(16))) _Float16 v16h;
typedef __attribute__((ext_vector_type(8)))  _Float16 v8h;
typedef __attribute__((ext_vector_type(4)))  _Float16 v4h;
typedef __attribute__((ext_vector_type(8)))  float    v8f;

#define D_MODEL 1024
#define N_HEADS 16
#define D_HEAD  64
#define SEQ     2048
#define BATCH   2
#define M_ROWS  (BATCH * SEQ)   // 4096

// ---- WMMA wrapper ----------------------------------------------------------
__device__ __forceinline__ v8f wmma_f16(v16h a, v16h b, v8f c) {
  // (neg_a, A, neg_b, B, c_mod, C, reuse_a, reuse_b)
  return __builtin_amdgcn_wmma_f32_16x16x32_f16(false, a, false, b,
                                                (short)0, c, false, false);
}

// ---- Fragment loaders (layouts per CDNA5 ISA §7.12.2) -----------
// A (16x32 f16, MxK): lane = {row = lane&15, half = lane>>4};
//   element j <-> K = half*8 + (j<8 ? j : j+8)  => two 16B chunks.
__device__ __forceinline__ v16h frag_a(const _Float16* A, int row0, int k0, int ld) {
  const int lane = threadIdx.x & 31;
  const int r = lane & 15, h = lane >> 4;
  const _Float16* p = A + (size_t)(row0 + r) * ld + k0 + h * 8;
  v8h c0 = *(const v8h*)(p);
  v8h c1 = *(const v8h*)(p + 16);
  v16h a;
#pragma unroll
  for (int j = 0; j < 8; ++j) { a[j] = c0[j]; a[j + 8] = c1[j]; }
  return a;
}

// B (32x16 f16, KxN), Bt stored N-major [N][K]: lane = {col = lane&15, half}:
//   element j <-> K = half*8 + (j<8 ? j : j+8)  => two 16B chunks.
__device__ __forceinline__ v16h frag_b(const _Float16* Bt, int col0, int k0, int ld) {
  const int lane = threadIdx.x & 31;
  const int c = lane & 15, h = lane >> 4;
  const _Float16* p = Bt + (size_t)(col0 + c) * ld + k0 + h * 8;
  v8h c0 = *(const v8h*)(p);
  v8h c1 = *(const v8h*)(p + 16);
  v16h b;
#pragma unroll
  for (int j = 0; j < 8; ++j) { b[j] = c0[j]; b[j + 8] = c1[j]; }
  return b;
}

// ---- Prep kernels ----------------------------------------------------------
__global__ void convert_f32_to_f16(const float* __restrict__ in,
                                   _Float16* __restrict__ out, int n4) {
  int i = blockIdx.x * blockDim.x + threadIdx.x;
  if (i >= n4) return;
  float4 v = ((const float4*)in)[i];
  v4h o; o[0] = (_Float16)v.x; o[1] = (_Float16)v.y;
         o[2] = (_Float16)v.z; o[3] = (_Float16)v.w;
  ((v4h*)out)[i] = o;
}

// W [K][N] f32 -> Wt [N][K] f16 (LDS-tiled transpose)
__global__ void transpose_w(const float* __restrict__ W, _Float16* __restrict__ Wt) {
  __shared__ float tile[32][33];
  int bx = blockIdx.x * 32, by = blockIdx.y * 32;
  int tx = threadIdx.x, ty = threadIdx.y;           // (32, 8)
#pragma unroll
  for (int i = 0; i < 32; i += 8)
    tile[ty + i][tx] = W[(size_t)(by + ty + i) * D_MODEL + bx + tx];
  __syncthreads();
#pragma unroll
  for (int i = 0; i < 32; i += 8)
    Wt[(size_t)(bx + ty + i) * D_MODEL + by + tx] = (_Float16)tile[tx][ty + i];
}

// ---- Fused QKV projection: wave computes 32x64 tile of x @ W ---------------
// z==0 -> Q [B,H,S,Dh], z==1 -> K [B,H,S,Dh], z==2 -> V^T [B,H,Dh,S]
__global__ void qkv_gemm(const _Float16* __restrict__ x16,
                         const _Float16* __restrict__ wqt,
                         const _Float16* __restrict__ wkt,
                         const _Float16* __restrict__ wvt,
                         const float* __restrict__ bq,
                         const float* __restrict__ bk,
                         const float* __restrict__ bv,
                         _Float16* __restrict__ q16,
                         _Float16* __restrict__ k16,
                         _Float16* __restrict__ vt16) {
  const int z = blockIdx.z;
  const _Float16* wt = (z == 0) ? wqt : (z == 1) ? wkt : wvt;
  const float* bias  = (z == 0) ? bq  : (z == 1) ? bk  : bv;
  _Float16* dst      = (z == 0) ? q16 : (z == 1) ? k16 : vt16;

  const int wid = blockIdx.x * (blockDim.x >> 5) + (threadIdx.x >> 5);
  const int m0 = (wid >> 4) * 32;      // 128 m-tiles
  const int n0 = (wid & 15) * 64;      // 16  n-tiles

  v8f acc[2][4];
#pragma unroll
  for (int i = 0; i < 2; ++i)
#pragma unroll
    for (int t = 0; t < 4; ++t) acc[i][t] = (v8f){};

  for (int kk = 0; kk < D_MODEL; kk += 32) {
    v16h a0 = frag_a(x16, m0,      kk, D_MODEL);
    v16h a1 = frag_a(x16, m0 + 16, kk, D_MODEL);
#pragma unroll
    for (int t = 0; t < 4; ++t) {
      v16h bf = frag_b(wt, n0 + t * 16, kk, D_MODEL);
      acc[0][t] = wmma_f16(a0, bf, acc[0][t]);
      acc[1][t] = wmma_f16(a1, bf, acc[1][t]);
    }
  }

  const int lane = threadIdx.x & 31;
  const int nl = lane & 15, hl = lane >> 4;
#pragma unroll
  for (int mi = 0; mi < 2; ++mi)
#pragma unroll
    for (int t = 0; t < 4; ++t) {
      const int col = n0 + t * 16 + nl;
      const int h = col >> 6, dh = col & 63;
      const float bv_ = bias[col];
#pragma unroll
      for (int r = 0; r < 8; ++r) {
        const int row = m0 + mi * 16 + r + hl * 8;
        const int b = row >> 11, s = row & (SEQ - 1);
        const float v = acc[mi][t][r] + bv_;
        if (z < 2)
          dst[(((size_t)(b * N_HEADS + h) * SEQ + s) * D_HEAD) + dh] = (_Float16)v;
        else
          dst[(((size_t)(b * N_HEADS + h) * D_HEAD + dh) * SEQ) + s] = (_Float16)v;
      }
    }
}

// ---- Flash attention: wave owns 16 q-rows of one (b,h) ---------------------
__global__ void attn_kernel(const _Float16* __restrict__ q16,
                            const _Float16* __restrict__ k16,
                            const _Float16* __restrict__ vt16,
                            const int*      __restrict__ mask,
                            const float*    __restrict__ pos_bias,
                            _Float16*       __restrict__ ctx16) {
  __shared__ __attribute__((aligned(16))) _Float16 pbuf[8][16 * 32];
  const int wave = threadIdx.x >> 5;
  const int lane = threadIdx.x & 31;
  const int wid = blockIdx.x * (blockDim.x >> 5) + wave;
  const int bh = wid >> 7;                 // 32 (b,h) pairs
  const int q0 = (wid & 127) * 16;         // 128 q-tiles
  const int b = bh >> 4, h = bh & 15;

  const _Float16* Q  = q16  + (size_t)bh * SEQ * D_HEAD;
  const _Float16* K  = k16  + (size_t)bh * SEQ * D_HEAD;
  const _Float16* Vt = vt16 + (size_t)bh * D_HEAD * SEQ;
  const float*    PB = pos_bias + (size_t)h * SEQ * SEQ;

  const int r = lane & 15, hl = lane >> 4;

  // Q A-fragments (Dh = 64 -> two K=32 fragments), reused across the key loop
  const v16h qa0 = frag_a(Q, q0, 0,  D_HEAD);
  const v16h qa1 = frag_a(Q, q0, 32, D_HEAD);

  v8f ctx[4];
#pragma unroll
  for (int t = 0; t < 4; ++t) ctx[t] = (v8f){};
  float m[8], l[8];
#pragma unroll
  for (int i = 0; i < 8; ++i) { m[i] = -__builtin_inff(); l[i] = 0.0f; }

  _Float16* P = pbuf[wave];
  const int kmax = q0 + 16;                // causal bound

  for (int kc = 0; kc < kmax; kc += 32) {
    const bool has2 = (kc + 16) < kmax;

    // scores: Q (16x64) @ K^T (64x32) as two 16x16 tiles
    v8f c0 = (v8f){}, c1 = (v8f){};
    {
      v16h kb0 = frag_b(K, kc, 0,  D_HEAD);
      v16h kb1 = frag_b(K, kc, 32, D_HEAD);
      c0 = wmma_f16(qa0, kb0, c0);
      c0 = wmma_f16(qa1, kb1, c0);
    }
    if (has2) {
      v16h kb2 = frag_b(K, kc + 16, 0,  D_HEAD);
      v16h kb3 = frag_b(K, kc + 16, 32, D_HEAD);
      c1 = wmma_f16(qa0, kb2, c1);
      c1 = wmma_f16(qa1, kb3, c1);
    }

    // prefetch next pos_bias chunk (global_prefetch_b8)
    if (kc + 32 < kmax)
      __builtin_prefetch(&PB[(size_t)(q0 + r) * SEQ + kc + 32 + hl * 16], 0, 1);

    // scale + pos_bias + mask
#pragma unroll
    for (int rr = 0; rr < 8; ++rr) {
      const int qrow = q0 + rr + hl * 8;
      const size_t rowoff = (size_t)qrow * SEQ;
      const int kk0 = kc + r;
      float s0 = c0[rr] * 0.125f + PB[rowoff + kk0];
      if (mask[rowoff + kk0] == 0) s0 = -1e9f;
      c0[rr] = s0;
      if (has2) {
        const int kk1 = kk0 + 16;
        float s1 = c1[rr] * 0.125f + PB[rowoff + kk1];
        if (mask[rowoff + kk1] == 0) s1 = -1e9f;
        c1[rr] = s1;
      } else {
        c1[rr] = -1e9f;
      }
    }

    // online softmax over the 32-col chunk; row lives across 16 lanes of a half
#pragma unroll
    for (int rr = 0; rr < 8; ++rr) {
      float mx = fmaxf(c0[rr], c1[rr]);
#pragma unroll
      for (int d = 1; d < 16; d <<= 1) mx = fmaxf(mx, __shfl_xor(mx, d, 32));
      const float mn = fmaxf(m[rr], mx);
      const float sc = __expf(m[rr] - mn);
      const float p0 = __expf(c0[rr] - mn);
      const float p1 = __expf(c1[rr] - mn);
      float rs = p0 + p1;
#pragma unroll
      for (int d = 1; d < 16; d <<= 1) rs += __shfl_xor(rs, d, 32);
      l[rr] = l[rr] * sc + rs;
      m[rr] = mn;
#pragma unroll
      for (int t = 0; t < 4; ++t) ctx[t][rr] = ctx[t][rr] * sc;
      // stage P in D-layout (row = rr + hl*8, col = r / r+16)
      P[(rr + hl * 8) * 32 + r]      = (_Float16)p0;
      P[(rr + hl * 8) * 32 + r + 16] = (_Float16)p1;
    }
    // LDS ops are in-order per wave; make the RAW explicit anyway
    asm volatile("s_wait_dscnt 0" ::: "memory");

    // ctx (16x64) += P (16x32) @ V (32x64); Vt is [Dh][S] so B cols contiguous
    const v16h pa = frag_a(P, 0, 0, 32);
#pragma unroll
    for (int t = 0; t < 4; ++t) {
      v16h vb = frag_b(Vt, t * 16, kc, SEQ);
      ctx[t] = wmma_f16(pa, vb, ctx[t]);
    }
  }

  // epilogue: normalize and store to ctx16 [B*S][D_MODEL] (heads concatenated)
#pragma unroll
  for (int t = 0; t < 4; ++t) {
    const int col = h * D_HEAD + t * 16 + r;
#pragma unroll
    for (int rr = 0; rr < 8; ++rr) {
      const int qrow = q0 + rr + hl * 8;
      const float v = ctx[t][rr] / l[rr];
      ctx16[((size_t)(b * SEQ + qrow)) * D_MODEL + col] = (_Float16)v;
    }
  }
}

// ---- Output projection: out = ctx @ Wo + bo (f32 out) ----------------------
__global__ void out_gemm(const _Float16* __restrict__ ctx16,
                         const _Float16* __restrict__ wot,
                         const float* __restrict__ bo,
                         float* __restrict__ out) {
  const int wid = blockIdx.x * (blockDim.x >> 5) + (threadIdx.x >> 5);
  const int m0 = (wid >> 4) * 32;
  const int n0 = (wid & 15) * 64;

  v8f acc[2][4];
#pragma unroll
  for (int i = 0; i < 2; ++i)
#pragma unroll
    for (int t = 0; t < 4; ++t) acc[i][t] = (v8f){};

  for (int kk = 0; kk < D_MODEL; kk += 32) {
    v16h a0 = frag_a(ctx16, m0,      kk, D_MODEL);
    v16h a1 = frag_a(ctx16, m0 + 16, kk, D_MODEL);
#pragma unroll
    for (int t = 0; t < 4; ++t) {
      v16h bf = frag_b(wot, n0 + t * 16, kk, D_MODEL);
      acc[0][t] = wmma_f16(a0, bf, acc[0][t]);
      acc[1][t] = wmma_f16(a1, bf, acc[1][t]);
    }
  }

  const int lane = threadIdx.x & 31;
  const int nl = lane & 15, hl = lane >> 4;
#pragma unroll
  for (int mi = 0; mi < 2; ++mi)
#pragma unroll
    for (int t = 0; t < 4; ++t) {
      const int col = n0 + t * 16 + nl;
      const float bv_ = bo[col];
#pragma unroll
      for (int r = 0; r < 8; ++r) {
        const int row = m0 + mi * 16 + r + hl * 8;
        out[(size_t)row * D_MODEL + col] = acc[mi][t][r] + bv_;
      }
    }
}

// ---------------------------------------------------------------------------
extern "C" void kernel_launch(void* const* d_in, const int* in_sizes, int n_in,
                              void* d_out, int out_size, void* d_ws, size_t ws_size,
                              hipStream_t stream) {
  const float* x        = (const float*)d_in[0];
  const int*   mask     = (const int*)  d_in[1];
  const float* Wq       = (const float*)d_in[2];
  const float* bq       = (const float*)d_in[3];
  const float* Wk       = (const float*)d_in[4];
  const float* bk       = (const float*)d_in[5];
  const float* Wv       = (const float*)d_in[6];
  const float* bv       = (const float*)d_in[7];
  const float* Wo       = (const float*)d_in[8];
  const float* bo       = (const float*)d_in[9];
  const float* pos_bias = (const float*)d_in[10];

  // workspace layout (f16 elements)
  _Float16* ws = (_Float16*)d_ws;
  size_t o = 0;
  _Float16* x16  = ws + o; o += (size_t)M_ROWS * D_MODEL;   // 8 MB
  _Float16* wqt  = ws + o; o += (size_t)D_MODEL * D_MODEL;  // 2 MB
  _Float16* wkt  = ws + o; o += (size_t)D_MODEL * D_MODEL;
  _Float16* wvt  = ws + o; o += (size_t)D_MODEL * D_MODEL;
  _Float16* wot  = ws + o; o += (size_t)D_MODEL * D_MODEL;
  _Float16* q16  = ws + o; o += (size_t)M_ROWS * D_MODEL;
  _Float16* k16  = ws + o; o += (size_t)M_ROWS * D_MODEL;
  _Float16* vt16 = ws + o; o += (size_t)M_ROWS * D_MODEL;
  _Float16* c16  = ws + o; o += (size_t)M_ROWS * D_MODEL;

  // 1) precision prep
  convert_f32_to_f16<<<(M_ROWS * D_MODEL / 4 + 255) / 256, 256, 0, stream>>>(
      x, x16, M_ROWS * D_MODEL / 4);
  dim3 tb(32, 8), tg(D_MODEL / 32, D_MODEL / 32);
  transpose_w<<<tg, tb, 0, stream>>>(Wq, wqt);
  transpose_w<<<tg, tb, 0, stream>>>(Wk, wkt);
  transpose_w<<<tg, tb, 0, stream>>>(Wv, wvt);
  transpose_w<<<tg, tb, 0, stream>>>(Wo, wot);

  // 2) fused QKV projection: 2048 waves per z, 8 waves per block
  qkv_gemm<<<dim3(256, 1, 3), 256, 0, stream>>>(x16, wqt, wkt, wvt,
                                                bq, bk, bv, q16, k16, vt16);

  // 3) flash attention: 32 (b,h) x 128 q-tiles = 4096 waves
  attn_kernel<<<dim3(512), 256, 0, stream>>>(q16, k16, vt16, mask, pos_bias, c16);

  // 4) output projection
  out_gemm<<<dim3(256), 256, 0, stream>>>(c16, wot, bo, (float*)d_out);
}
